// ShiftConv3d_v3_80900003988109
// MI455X (gfx1250) — matrix operand order repaired and checked by reading, not compiled, not run on hardware
//
#include <hip/hip_runtime.h>

// ---------------------------------------------------------------------------
// ShiftConv3d for MI455X (gfx1250), wave32 + WMMA + TDM.
// fp32 conv is compute-bound (1.3e11 FLOP vs ~7us HBM floor), so use bf16
// hi/lo split: 3x v_wmma_f32_16x16x32_bf16 per K=32 chunk (fp32-grade accuracy,
// ~2.7x native fp32-WMMA rate). Implicit GEMM per WG: M=128 out-ch x N=96 px
// (one output row), K=9 taps x 128 ch. Per-tap 64KB weight tiles are streamed
// into LDS by the Tensor Data Mover, double-buffered to overlap with WMMA.
// ---------------------------------------------------------------------------

typedef __bf16 bf16_t;
typedef __attribute__((ext_vector_type(8)))  __bf16 v8bf;
typedef __attribute__((ext_vector_type(16))) __bf16 v16bf;
typedef __attribute__((ext_vector_type(8)))  float  v8f;
typedef unsigned int u32x4 __attribute__((ext_vector_type(4)));
typedef int i32x4 __attribute__((ext_vector_type(4)));
typedef int i32x8 __attribute__((ext_vector_type(8)));

static constexpr int CCH   = 128;
static constexpr int HH    = 96;
static constexpr int WWD   = 96;
static constexpr int TOUT  = 8;
static constexpr int NB    = 2;
static constexpr int WTAP  = CCH * CCH;               // 16384 weights per tap per conv
static constexpr int WELEMS = 3 * 9 * WTAP;           // 442,368 (per plane)
static constexpr long long TELEMS = (long long)NB * CCH * TOUT * HH * WWD; // 18,874,368
static constexpr long long OSTRIDE = (long long)TOUT * HH * WWD;           // 73,728
static constexpr int CPAD  = 136;                     // padded channel stride in LDS
static constexpr int INP   = 3 * 98 * CPAD;           // bf16 elems per input plane
// LDS: 2 input planes + 2 double-buffered [hi|lo] weight tiles
static constexpr int SMEM_BYTES = (2 * INP + 4 * WTAP) * 2;  // 291,008 B

#define WMMA_BF16(A, B, Cc) \
  __builtin_amdgcn_wmma_f32_16x16x32_bf16(false, (A), false, (B), (short)0, (Cc), false, false)

// ---- shift offset tables (from _offsets: element 0 and 2 of each base triple)
__device__ __constant__ int g_B0[11] = {-2,-1,-1,-1, 1, 1, 1, 2, 0, 0, 0};
__device__ __constant__ int g_B2[11] = { 1, 0, 1, 2, 0,-1,-2,-1,-1, 0, 1};

__device__ __forceinline__ int iclamp(int v, int lo, int hi) {
  return v < lo ? lo : (v > hi ? hi : v);
}

// A-operand: 16-bit A 16x32 layout. lane L: M=L%16, khalf=L/16.
// elems 0..15 = K (khalf*16 .. +15) contiguous, matching load_b.
__device__ __forceinline__ v16bf load_a(const bf16_t* wb, int o, int c0, int khalf) {
  const v8bf a0 = *(const v8bf*)(wb + o * 128 + c0 + khalf * 16);
  const v8bf a1 = *(const v8bf*)(wb + o * 128 + c0 + 8 + khalf * 16);
  return __builtin_shufflevector(a0, a1, 0,1,2,3,4,5,6,7,8,9,10,11,12,13,14,15);
}

// B-operand: 32x16 in this code's own k order (same as load_a). lane L: N=L%16, K = (L/16)*16 + 0..15 contiguous.
__device__ __forceinline__ v16bf load_b(const bf16_t* p) {
  const v8bf x0 = *(const v8bf*)p;
  const v8bf x1 = *(const v8bf*)(p + 8);
  return __builtin_shufflevector(x0, x1, 0,1,2,3,4,5,6,7,8,9,10,11,12,13,14,15);
}

// ---------------------------------------------------------------------------
// TDM: DMA one contiguous 64KB weight tile (hi+lo planes) global -> LDS.
// D#: 1-D tile, data_size=8B, tile_dim0=tensor_dim0=8192, dim1=1.
// 6-arg builtin form (clang-23 / therock-10.0): (g0, g1, g2, g3, g4, cpol).
// ---------------------------------------------------------------------------
__device__ __forceinline__ void tdm_load_weights(const bf16_t* gsrc, bf16_t* ldst) {
  unsigned lds = (unsigned)(uintptr_t)(void*)ldst;           // LDS byte offset
  unsigned long long ga = (unsigned long long)(uintptr_t)gsrc;
  u32x4 g0;
  g0[0] = 1u;                                  // count=1, user descriptor
  g0[1] = lds;                                 // lds_addr
  g0[2] = (unsigned)ga;                        // global_addr[31:0]
  g0[3] = (unsigned)(ga >> 32) | (2u << 30);   // global_addr[56:32] | type=2
  i32x8 g1;
  g1[0] = (int)(3u << 16);                     // wg_mask=0, data_size=8B
  g1[1] = (int)(8192u << 16);                  // tensor_dim0[15:0] in [31:16]
  g1[2] = (int)(1u << 16);                     // tensor_dim0 hi=0, tensor_dim1 lo=1
  g1[3] = (int)(8192u << 16);                  // tensor_dim1 hi=0, tile_dim0=8192
  g1[4] = 1;                                   // tile_dim1=1, tile_dim2=0
  g1[5] = 8192;                                // tensor_dim0_stride[31:0]
  g1[6] = 0;
  g1[7] = 0;
  i32x4 z4 = {0, 0, 0, 0};                     // groups 2/3 unused (<=2D)
  i32x8 z8 = {0, 0, 0, 0, 0, 0, 0, 0};
  __builtin_amdgcn_tensor_load_to_lds(g0, g1, z4, z4, z8, 0);
}

// ---------------------------------------------------------------------------
// Kernel 0: split fp32 weights into bf16 hi/lo, layout [l][tap][hi|lo][o][c]
// so each (l,tap) is one contiguous 64KB block for the TDM.
// ---------------------------------------------------------------------------
__global__ void sc3d_split_weights(const float* __restrict__ w1,
                                   const float* __restrict__ w2,
                                   const float* __restrict__ w3,
                                   bf16_t* __restrict__ wsplit) {
  int idx = blockIdx.x * 256 + threadIdx.x;
  if (idx >= WELEMS) return;
  int l = idx / (CCH * CCH * 9);
  int r = idx % (CCH * CCH * 9);
  const float* src = (l == 0) ? w1 : ((l == 1) ? w2 : w3);
  float v = src[r];                       // src layout [o][c][tap]
  int o = r / 1152, r2 = r % 1152, c = r2 / 9, tap = r2 % 9;
  bf16_t hi = (bf16_t)v;
  bf16_t lo = (bf16_t)(v - (float)hi);
  int base = (l * 9 + tap) * 2 * WTAP + o * 128 + c;
  wsplit[base] = hi;
  wsplit[base + WTAP] = lo;
}

// ---------------------------------------------------------------------------
// Kernel 1: implicit-GEMM 1x3x3 conv, bf16x3 split, WMMA 16x16x32, TDM weights.
// grid = (96 rows, 16 frames, 3 convs); 256 threads = 8 waves (4 M x 2 N).
// ---------------------------------------------------------------------------
__global__ __launch_bounds__(256) void sc3d_conv_wmma(
    const float* __restrict__ x,
    const float* __restrict__ b1, const float* __restrict__ b2,
    const float* __restrict__ b3,
    const bf16_t* __restrict__ wsplit_g,
    float* __restrict__ tout) {
  extern __shared__ char smem_raw[];
  bf16_t* in_hi = (bf16_t*)smem_raw;       // [3 rows][98 px][CPAD ch]
  bf16_t* in_lo = in_hi + INP;
  bf16_t* wbuf0 = in_lo + INP;             // [hi:128x128 | lo:128x128]
  bf16_t* wbuf1 = wbuf0 + 2 * WTAP;

  const int h  = blockIdx.x;               // output row 0..95
  const int bt = blockIdx.y;               // frame 0..15
  const int l  = blockIdx.z;               // conv 0..2
  const int b  = bt >> 3, t = bt & 7;
  const int tin = t + l;                   // input temporal frame
  const int tid = threadIdx.x;
  const int lane = tid & 31, wv = tid >> 5;
  const int mbase = (wv & 3) * 32;         // wave M base (out-channels)
  const int nbase = (wv >> 2) * 48;        // wave N base (pixels)
  const int khalf = lane >> 4, lmod = lane & 15;

  const bf16_t* wtap_g = wsplit_g + (size_t)l * 9 * 2 * WTAP;

  // kick off tap-0 weight DMA; input staging below overlaps it
  if (wv == 0) tdm_load_weights(wtap_g, wbuf0);

  // ---- stage input rows h-1..h+1 (zero-padded), fp32 -> bf16 hi/lo, transpose
  for (int i = tid; i < 3 * 98 * 128; i += 256) {
    int c = i / 294, rem = i % 294, r = rem / 98, px = rem % 98;
    int row = h + r - 1, wc = px - 1;
    float v = 0.0f;
    if ((unsigned)row < 96u && (unsigned)wc < 96u)
      v = x[(((size_t)(b * 128 + c) * 10 + tin) * 96 + row) * 96 + wc];
    bf16_t hi = (bf16_t)v;
    bf16_t lo = (bf16_t)(v - (float)hi);
    int o = (r * 98 + px) * CPAD + c;
    in_hi[o] = hi;
    in_lo[o] = lo;
  }

  v8f acc[2][3] = {};

#pragma unroll 1
  for (int tap = 0; tap < 9; ++tap) {
    if (wv == 0) __builtin_amdgcn_s_wait_tensorcnt(0);  // tap's tile landed
    __syncthreads();                                    // publish to all waves
    // prefetch next tap into the other buffer (last read in the previous tap,
    // protected by the barrier above) -- overlaps with this tap's WMMAs
    if (wv == 0 && tap + 1 < 9)
      tdm_load_weights(wtap_g + (size_t)(tap + 1) * 2 * WTAP,
                       (tap & 1) ? wbuf0 : wbuf1);

    const bf16_t* w_hi = (tap & 1) ? wbuf1 : wbuf0;
    const bf16_t* w_lo = w_hi + WTAP;
    const int kh = tap / 3, kw = tap % 3;
#pragma unroll
    for (int c0 = 0; c0 < 128; c0 += 32) {
      v16bf ah[2], al[2], bh[3], bl[3];
#pragma unroll
      for (int ms = 0; ms < 2; ++ms) {
        int o = mbase + ms * 16 + lmod;
        ah[ms] = load_a(w_hi, o, c0, khalf);
        al[ms] = load_a(w_lo, o, c0, khalf);
      }
#pragma unroll
      for (int ns = 0; ns < 3; ++ns) {
        int px = nbase + ns * 16 + lmod + kw;           // col = w + kw - 1, +1 pad
        int off = (kh * 98 + px) * CPAD + c0 + khalf * 16;
        bh[ns] = load_b(in_hi + off);
        bl[ns] = load_b(in_lo + off);
      }
#pragma unroll
      for (int ms = 0; ms < 2; ++ms)
#pragma unroll
        for (int ns = 0; ns < 3; ++ns) {
          acc[ms][ns] = WMMA_BF16(ah[ms], bh[ns], acc[ms][ns]);  // hi*hi
          acc[ms][ns] = WMMA_BF16(ah[ms], bl[ns], acc[ms][ns]);  // hi*lo
          acc[ms][ns] = WMMA_BF16(al[ms], bh[ns], acc[ms][ns]);  // lo*hi
        }
    }
  }

  // ---- epilogue: add bias, store fp32 to t-buffer [b][o][t][h][w]
  const float* bias = (l == 0) ? b1 : ((l == 1) ? b2 : b3);
  float* tb = tout + (size_t)l * (size_t)TELEMS;
  const size_t fbase = (size_t)b * (size_t)(128 * OSTRIDE) +
                       (size_t)t * (size_t)(HH * WWD) + (size_t)h * WWD;
#pragma unroll
  for (int ms = 0; ms < 2; ++ms) {
    int obase = mbase + ms * 16 + khalf * 8;  // D layout: lanes16-31 hold M=8..15
#pragma unroll
    for (int ns = 0; ns < 3; ++ns) {
      int wcol = nbase + ns * 16 + lmod;      // N = lane%16
#pragma unroll
      for (int r = 0; r < 8; ++r) {
        int o = obase + r;
        tb[fbase + (size_t)o * (size_t)OSTRIDE + wcol] = acc[ms][ns][r] + bias[o];
      }
    }
  }
}

// ---------------------------------------------------------------------------
// Kernel 2: per-channel edge-clamped shift gather of t1/t3 + t2, sum to out.
// ---------------------------------------------------------------------------
__global__ void sc3d_shift_sum(const float* __restrict__ tbuf,
                               float* __restrict__ out) {
  const float* t1 = tbuf;
  const float* t2 = tbuf + TELEMS;
  const float* t3 = tbuf + 2 * TELEMS;
  long long idx = (long long)blockIdx.x * 256 + threadIdx.x;
  if (idx >= TELEMS) return;
  int w = (int)(idx % 96);
  int h = (int)((idx / 96) % 96);
  int o = (int)((idx / (96 * 96 * 8)) % 128);
  long long fb = idx - (long long)h * 96 - w;   // base of (b,o,t) spatial plane

  int dh1 = 0, dw1 = 0, dh3 = 0, dw3 = 0;
  if (o < 121) {
    int i = o / 11, j = o % 11;
    dh1 = g_B0[i]; dw1 = g_B0[j];
    dh3 = g_B2[i]; dw3 = g_B2[j];
  }
  int h1 = iclamp(h - dh1, 0, 95), w1 = iclamp(w - dw1, 0, 95);
  int h3 = iclamp(h - dh3, 0, 95), w3 = iclamp(w - dw3, 0, 95);

  out[idx] = t1[fb + (long long)h1 * 96 + w1] + t2[idx] +
             t3[fb + (long long)h3 * 96 + w3];
}

// ---------------------------------------------------------------------------
extern "C" void kernel_launch(void* const* d_in, const int* in_sizes, int n_in,
                              void* d_out, int out_size, void* d_ws, size_t ws_size,
                              hipStream_t stream) {
  const float* x  = (const float*)d_in[0];
  const float* w1 = (const float*)d_in[1];
  const float* b1 = (const float*)d_in[2];
  const float* w2 = (const float*)d_in[3];
  const float* b2 = (const float*)d_in[4];
  const float* w3 = (const float*)d_in[5];
  const float* b3 = (const float*)d_in[6];
  float* out = (float*)d_out;

  // workspace layout: [wsplit (1.77MB) | t1 | t2 | t3]  (~228 MB total)
  char* ws = (char*)d_ws;
  const size_t WBYTES = (size_t)WELEMS * 2 * 2;    // hi+lo, 1,769,472 B
  bf16_t* wsplit = (bf16_t*)ws;
  float* tbuf = (float*)(ws + WBYTES);

  sc3d_split_weights<<<(WELEMS + 255) / 256, 256, 0, stream>>>(w1, w2, w3, wsplit);

  sc3d_conv_wmma<<<dim3(96, 16, 3), 256, SMEM_BYTES, stream>>>(
      x, b1, b2, b3, wsplit, tbuf);

  sc3d_shift_sum<<<(unsigned)((TELEMS + 255) / 256), 256, 0, stream>>>(tbuf, out);
}
